// PowerLawAttention_72267119723248
// MI455X (gfx1250) — matrix operand order repaired and checked by reading, not compiled, not run on hardware
//
#include <hip/hip_runtime.h>
#include <hip/hip_bf16.h>

// ---------------- problem constants (from reference) ----------------
#define D_MODEL 1024
#define HEADS   16
#define D_K     64
#define BSZ     4
#define LQ      512
#define LK      2048

typedef __attribute__((ext_vector_type(16))) _Float16 v16h;
typedef __attribute__((ext_vector_type(8)))  float    v8f;

#if __has_builtin(__builtin_amdgcn_tensor_load_to_lds)
#define USE_TDM 1
#else
#define USE_TDM 0
#endif

#if USE_TDM
typedef __attribute__((ext_vector_type(4))) unsigned tdm_v4u;
typedef __attribute__((ext_vector_type(8))) int      tdm_v8i;
typedef __attribute__((ext_vector_type(4))) int      tdm_v4i;

// TDM DMA of one 32x64 _Float16 tile (row stride 64 elems in memory) into
// LDS, padded to 72 halfs per row (pad_interval = 32 DWORDs, pad_amount =
// 4 DWORDs). Descriptor per CDNA5 ISA.
__device__ __forceinline__ void tdm_load_tile_f16(const _Float16* gsrc,
                                                  unsigned lds_byte_off) {
    const unsigned long long ga = (unsigned long long)(size_t)gsrc;
    tdm_v4u g0;
    g0[0] = 1u;                                   // count=1, user descriptor
    g0[1] = lds_byte_off;                         // lds_addr
    g0[2] = (unsigned)ga;                         // global_addr[31:0]
    g0[3] = (unsigned)((ga >> 32) & 0x01FFFFFFu)  // global_addr[56:32]
            | (2u << 30);                         // type = 2 (image)
    tdm_v8i g1;
    g1[0] = (1 << 16)                             // data_size = 2 bytes
          | (1 << 20)                             // pad_enable
          | (4 << 22)                             // pad_interval = 32 DWORDs
          | (3 << 25);                            // pad_amount  = 4 DWORDs
    g1[1] = (int)(64u << 16);                     // tensor_dim0 = 64
    g1[2] = (int)(32u << 16);                     // tensor_dim1 = 32
    g1[3] = (int)(64u << 16);                     // tile_dim0 = 64
    g1[4] = 32;                                   // tile_dim1 = 32, tile_dim2 = 0
    g1[5] = 64;                                   // tensor_dim0_stride = 64
    g1[6] = 0;
    g1[7] = 0;
    tdm_v4i gz = {0, 0, 0, 0};
#if __clang_major__ >= 23
    tdm_v8i gz8 = {0, 0, 0, 0, 0, 0, 0, 0};
    __builtin_amdgcn_tensor_load_to_lds(g0, g1, gz, gz, gz8, 0);
#else
    __builtin_amdgcn_tensor_load_to_lds(g0, g1, gz, gz, 0);
#endif
}
#endif  // USE_TDM

// ====================================================================
// Generic tiled WMMA GEMM:  C[M,N] = A[M,K] @ B[N,K]^T + bias[N]
//   A_F16:    A is _Float16 (else fp32, converted while staging)
//   HEAD_OUT: scatter output as f16 into head-major (b,h,L,64) layout
//             (col = h*64+d, row = b*L+l), else fp32 row-major.
// One 256-thread block (8 wave32) computes a 64x64 tile; each wave owns
// two 16x16 WMMA accumulators; K is consumed 32 at a time through LDS.
// ====================================================================
template <bool A_F16, bool HEAD_OUT>
__global__ __launch_bounds__(256) void gemm_bias_wmma(
    const void* __restrict__ Av, const float* __restrict__ B,
    const float* __restrict__ bias, void* __restrict__ Out,
    int M, int N, int K, int L)
{
    __shared__ _Float16 As[64][40];   // 64 rows x 32 k (pad to 40)
    __shared__ _Float16 Bs[64][40];

    const int tid  = threadIdx.x;
    const int lane = tid & 31;
    const int wave = tid >> 5;
    const int grp  = lane >> 4;       // 0 or 1 (wave32 half)
    const int ln   = lane & 15;
    const int m0   = blockIdx.y * 64;
    const int n0   = blockIdx.x * 64;
    const int rt   = wave & 3;        // row tile 0..3
    const int ct0  = wave >> 2;       // col tiles ct0 and ct0+2

    v8f acc0 = {}, acc1 = {};

    for (int kb = 0; kb < K; kb += 32) {
        __syncthreads();
        // ---- stage A (64x32) and B (64x32) as f16 ----
        {
            const int row = tid >> 2;          // 0..63
            const int c8  = (tid & 3) * 8;     // 0,8,16,24
            if constexpr (A_F16) {
                const _Float16* Af = (const _Float16*)Av + (size_t)(m0 + row) * K + kb + c8;
                #pragma unroll
                for (int i = 0; i < 8; ++i) As[row][c8 + i] = Af[i];
            } else {
                const float* Af = (const float*)Av + (size_t)(m0 + row) * K + kb + c8;
                #pragma unroll
                for (int i = 0; i < 8; ++i) As[row][c8 + i] = (_Float16)Af[i];
            }
            const float* Bf = B + (size_t)(n0 + row) * K + kb + c8;
            #pragma unroll
            for (int i = 0; i < 8; ++i) Bs[row][c8 + i] = (_Float16)Bf[i];
        }
        __syncthreads();

        // ---- A fragment: 16x32, lane ln = row, halves select k-range ----
        v16h af;
        {
            const int m = rt * 16 + ln;
            #pragma unroll
            for (int i = 0; i < 8; ++i) {
                af[i]     = As[m][grp * 16 + i];
                af[8 + i] = As[m][grp * 16 + 8 + i];
            }
        }
        // ---- B fragments: 32x16, lane ln = col, grp selects k-half ----
        v16h bf0, bf1;
        #pragma unroll
        for (int i = 0; i < 16; ++i) bf0[i] = Bs[ct0 * 16 + ln][grp * 16 + i];
        #pragma unroll
        for (int i = 0; i < 16; ++i) bf1[i] = Bs[(ct0 + 2) * 16 + ln][grp * 16 + i];

        acc0 = __builtin_amdgcn_wmma_f32_16x16x32_f16(false, af, false, bf0,
                                                      (short)0, acc0, false, false);
        acc1 = __builtin_amdgcn_wmma_f32_16x16x32_f16(false, af, false, bf1,
                                                      (short)0, acc1, false, false);
    }

    // ---- epilogue: bias add + scatter ----
    #pragma unroll
    for (int t = 0; t < 2; ++t) {
        const v8f acc = t ? acc1 : acc0;
        const int ct  = ct0 + 2 * t;
        const int col = n0 + ct * 16 + ln;
        const float bv = bias[col];
        #pragma unroll
        for (int r = 0; r < 8; ++r) {
            const int row = m0 + rt * 16 + r + 8 * grp;
            const float v = acc[r] + bv;
            if constexpr (HEAD_OUT) {
                const int b = row / L, l = row % L;
                const int h = col >> 6, d = col & 63;
                ((_Float16*)Out)[((((size_t)b * HEADS + h) * L + l) << 6) + d] = (_Float16)v;
            } else {
                ((float*)Out)[(size_t)row * N + col] = v;
            }
        }
    }
}

// ====================================================================
// Flash attention with learned additive bias + mask.
// grid = (Lq/128, bs*HEADS); block = 256 (8 waves x 16 query rows).
// K/V chunks (32 rows x 64 d, f16) are double-buffered in LDS and moved
// by the Tensor Data Mover: wave 0 issues tensor_load_to_lds for chunk
// i+1, waits s_wait_tensorcnt<=2 for chunk i, barrier releases block.
// Per chunk: scores via wmma f16 (2 tiles x 2 d-chunks), s/8 + bias -
// mask, online softmax, P->LDS transpose, O += P@V (4 wmma).
// ====================================================================
__global__ __launch_bounds__(256) void flash_attn_bias(
    const _Float16* __restrict__ Qh, const _Float16* __restrict__ Kh,
    const _Float16* __restrict__ Vh, const int* __restrict__ b_idx,
    const int* __restrict__ mask, const float* __restrict__ b_table,
    _Float16* __restrict__ Ctx)
{
    __shared__ _Float16 Ks[2][32][72];   // double-buffered 32 k-rows x 64 d (pad 72)
    __shared__ _Float16 Vs[2][32][72];
    __shared__ _Float16 Ps[8][16][40];   // per-wave P transpose buffer

    const int tid  = threadIdx.x;
    const int lane = tid & 31;
    const int wave = tid >> 5;
    const int grp  = lane >> 4;
    const int ln   = lane & 15;
    const int bh   = blockIdx.y;             // b*HEADS + h
    const int b    = bh >> 4;
    const int h    = bh & 15;
    const int q0   = blockIdx.x * 128 + wave * 16;

    // ---- preload Q fragments (16x32 per d-chunk), loop-invariant ----
    const _Float16* qbase = Qh + ((size_t)bh * LQ + q0) * D_K;
    v16h qf[2];
    #pragma unroll
    for (int c = 0; c < 2; ++c)
        #pragma unroll
        for (int i = 0; i < 8; ++i) {
            qf[c][i]     = qbase[(size_t)ln * D_K + c * 32 + grp * 16 + i];
            qf[c][8 + i] = qbase[(size_t)ln * D_K + c * 32 + grp * 16 + 8 + i];
        }

    float rowM[8], rowL[8];
    #pragma unroll
    for (int r = 0; r < 8; ++r) { rowM[r] = -1e30f; rowL[r] = 0.f; }
    v8f acc[4] = {v8f{}, v8f{}, v8f{}, v8f{}};

    const _Float16* kbase = Kh + (size_t)bh * LK * D_K;
    const _Float16* vbase = Vh + (size_t)bh * LK * D_K;

#if USE_TDM
    if (tid == 0) {   // DMA chunk 0 into buffer 0 (EXEC ignored by TDM)
        tdm_load_tile_f16(kbase, (unsigned)(size_t)&Ks[0][0][0]);
        tdm_load_tile_f16(vbase, (unsigned)(size_t)&Vs[0][0][0]);
    }
#endif

    for (int kc = 0; kc < LK; kc += 32) {
        const int cur = (kc >> 5) & 1;
#if USE_TDM
        __syncthreads();   // prefetch target buffer no longer being read
        if (tid == 0) {
            if (kc + 32 < LK) {
                tdm_load_tile_f16(kbase + (size_t)(kc + 32) * D_K,
                                  (unsigned)(size_t)&Ks[1 - cur][0][0]);
                tdm_load_tile_f16(vbase + (size_t)(kc + 32) * D_K,
                                  (unsigned)(size_t)&Vs[1 - cur][0][0]);
                __builtin_amdgcn_s_wait_tensorcnt(2);  // current pair done
            } else {
                __builtin_amdgcn_s_wait_tensorcnt(0);
            }
        }
        __syncthreads();   // current buffer ready for all waves
#else
        __syncthreads();
        {   // manual staging fallback: 32 rows x 64 halfs each
            const int row = tid >> 3;        // 0..31
            const int c8  = (tid & 7) * 8;   // 0..56
            const _Float16* kp = kbase + (size_t)(kc + row) * D_K + c8;
            const _Float16* vp = vbase + (size_t)(kc + row) * D_K + c8;
            __builtin_prefetch(kp + 32 * D_K, 0, 0);
            __builtin_prefetch(vp + 32 * D_K, 0, 0);
            #pragma unroll
            for (int i = 0; i < 8; ++i) Ks[cur][row][c8 + i] = kp[i];
            #pragma unroll
            for (int i = 0; i < 8; ++i) Vs[cur][row][c8 + i] = vp[i];
        }
        __syncthreads();
#endif

        // ---- scores: two 16x16 tiles over this 32-col chunk ----
        v8f s[2] = {v8f{}, v8f{}};
        #pragma unroll
        for (int t = 0; t < 2; ++t)
            #pragma unroll
            for (int c = 0; c < 2; ++c) {
                v16h kf;
                #pragma unroll
                for (int i = 0; i < 16; ++i)
                    kf[i] = Ks[cur][t * 16 + ln][c * 32 + grp * 16 + i];
                s[t] = __builtin_amdgcn_wmma_f32_16x16x32_f16(false, qf[c], false, kf,
                                                              (short)0, s[t], false, false);
            }

        // ---- scale + learned bias + mask ----
        float tv[2][8];
        #pragma unroll
        for (int t = 0; t < 2; ++t) {
            const int kcol = kc + t * 16 + ln;
            #pragma unroll
            for (int r = 0; r < 8; ++r) {
                const size_t off = ((size_t)b * LQ + (q0 + r + 8 * grp)) * (size_t)LK + kcol;
                const int bi = b_idx[off];
                const int mk = mask[off];
                const float val = s[t][r] * 0.125f + b_table[bi * HEADS + h];
                tv[t][r] = (mk == 0) ? -1e9f : val;
            }
        }

        // ---- online softmax (row stats across 16-lane half-wave) ----
        float alpha[8];
        #pragma unroll
        for (int r = 0; r < 8; ++r) {
            float mx = fmaxf(tv[0][r], tv[1][r]);
            #pragma unroll
            for (int o = 8; o >= 1; o >>= 1) mx = fmaxf(mx, __shfl_xor(mx, o, 32));
            const float mnew = fmaxf(rowM[r], mx);
            alpha[r] = __expf(rowM[r] - mnew);
            const float p0 = __expf(tv[0][r] - mnew);
            const float p1 = __expf(tv[1][r] - mnew);
            tv[0][r] = p0; tv[1][r] = p1;
            float sm = p0 + p1;
            #pragma unroll
            for (int o = 8; o >= 1; o >>= 1) sm += __shfl_xor(sm, o, 32);
            rowL[r] = rowL[r] * alpha[r] + sm;
            rowM[r] = mnew;
        }

        // ---- P: C-fragment layout -> LDS -> A-fragment layout ----
        #pragma unroll
        for (int t = 0; t < 2; ++t)
            #pragma unroll
            for (int r = 0; r < 8; ++r)
                Ps[wave][r + 8 * grp][t * 16 + ln] = (_Float16)tv[t][r];
        asm volatile("s_wait_dscnt 0" ::: "memory");   // intra-wave LDS RAW
        v16h pf;
        #pragma unroll
        for (int i = 0; i < 8; ++i) {
            pf[i]     = Ps[wave][ln][grp * 16 + i];
            pf[8 + i] = Ps[wave][ln][grp * 16 + 8 + i];
        }

        // ---- O = O*alpha + P @ V  (4 column tiles of d) ----
        #pragma unroll
        for (int dt = 0; dt < 4; ++dt) {
            v16h vf;
            #pragma unroll
            for (int i = 0; i < 16; ++i)
                vf[i] = Vs[cur][grp * 16 + i][dt * 16 + ln];
            v8f a = acc[dt];
            #pragma unroll
            for (int r = 0; r < 8; ++r) a[r] *= alpha[r];
            acc[dt] = __builtin_amdgcn_wmma_f32_16x16x32_f16(false, pf, false, vf,
                                                             (short)0, a, false, false);
        }
    }

    // ---- normalize and write context (b, q, h*64+d) as f16 ----
    #pragma unroll
    for (int dt = 0; dt < 4; ++dt)
        #pragma unroll
        for (int r = 0; r < 8; ++r) {
            const int q = q0 + r + 8 * grp;
            const float v = acc[dt][r] / rowL[r];
            Ctx[((size_t)b * LQ + q) * D_MODEL + h * D_K + dt * 16 + ln] = (_Float16)v;
        }
}

// ====================================================================
// Host-side orchestration (graph-capture safe: launches only).
// inputs: 0=q 1=k 2=v 3=b_idx 4=mask 5=Wq 6=bq 7=Wk 8=bk 9=Wv 10=bv
//         11=Wo 12=bo 13=b_table
// workspace: qh | kh | vh | ctx (all _Float16)
// ====================================================================
extern "C" void kernel_launch(void* const* d_in, const int* in_sizes, int n_in,
                              void* d_out, int out_size, void* d_ws, size_t ws_size,
                              hipStream_t stream) {
    (void)in_sizes; (void)n_in; (void)out_size; (void)ws_size;

    _Float16* qh  = (_Float16*)d_ws;
    _Float16* kh  = qh + (size_t)BSZ * HEADS * LQ * D_K;   // 2M halfs
    _Float16* vh  = kh + (size_t)BSZ * HEADS * LK * D_K;   // +8M halfs
    _Float16* ctx = vh + (size_t)BSZ * HEADS * LK * D_K;   // +8M halfs

    const dim3 blk(256);

    // Q/K/V projections: fp32 in, f16 head-major out
    gemm_bias_wmma<false, true><<<dim3(D_MODEL / 64, (BSZ * LQ) / 64), blk, 0, stream>>>(
        d_in[0], (const float*)d_in[5], (const float*)d_in[6], qh,
        BSZ * LQ, D_MODEL, D_MODEL, LQ);
    gemm_bias_wmma<false, true><<<dim3(D_MODEL / 64, (BSZ * LK) / 64), blk, 0, stream>>>(
        d_in[1], (const float*)d_in[7], (const float*)d_in[8], kh,
        BSZ * LK, D_MODEL, D_MODEL, LK);
    gemm_bias_wmma<false, true><<<dim3(D_MODEL / 64, (BSZ * LK) / 64), blk, 0, stream>>>(
        d_in[2], (const float*)d_in[9], (const float*)d_in[10], vh,
        BSZ * LK, D_MODEL, D_MODEL, LK);

    // fused biased attention + softmax + PV (TDM double-buffered K/V)
    flash_attn_bias<<<dim3(LQ / 128, BSZ * HEADS), blk, 0, stream>>>(
        qh, kh, vh, (const int*)d_in[3], (const int*)d_in[4],
        (const float*)d_in[13], ctx);

    // output projection: f16 in, fp32 out
    gemm_bias_wmma<true, false><<<dim3(D_MODEL / 64, (BSZ * LQ) / 64), blk, 0, stream>>>(
        ctx, (const float*)d_in[11], (const float*)d_in[12], d_out,
        BSZ * LQ, D_MODEL, D_MODEL, 1);
}
